// EdgeClassifierGNN_16552803959009
// MI455X (gfx1250) — compile-verified, matrix operand order checked by reading, not run on hardware
//
#include <hip/hip_runtime.h>
#include <hip/hip_bf16.h>

typedef __attribute__((ext_vector_type(16))) _Float16 v16h;
typedef __attribute__((ext_vector_type(8)))  float    v8f;

#define NNODES 50000
#define NEDGES 800000
#define IN_DIMC 128
#define HIDC    64
#define OUTC    2

// ---------------------------------------------------------------------------
// Utility: zero a float region (grid-stride)
// ---------------------------------------------------------------------------
__global__ void zero_kernel(float* __restrict__ p, long n) {
    long i = (long)blockIdx.x * blockDim.x + threadIdx.x;
    long stride = (long)gridDim.x * blockDim.x;
    for (; i < n; i += stride) p[i] = 0.0f;
}

// ---------------------------------------------------------------------------
// Pre-swizzle weights into exact WMMA B-fragment lane order (f16).
// Logical weight is [2*half x 64]: rows [0,half) from srcA, rows [half,2*half)
// from srcB.  Fragment (s,t): K rows s*32..s*32+31, N cols t*16..t*16+15.
// Lane l (half-wave h=l>>4, lrow=l&15) owns column n=t*16+lrow and elements
// e<8 -> k=s*32+h*8+e ; e>=8 -> k=s*32+h*8+8+e   (ISA 7.12.2, 16-bit B 32x16).
// Output: dst[frag*32*16 + lane*16 + e]  -> one aligned 32B v16h per lane.
// ---------------------------------------------------------------------------
__global__ void swz_weights_kernel(const float* __restrict__ srcA, const float* __restrict__ srcB,
                                   _Float16* __restrict__ dst, int half, int nsteps) {
    int i = blockIdx.x * blockDim.x + threadIdx.x;    // frag*32 + lane
    int nfrag = nsteps * 4;
    if (i >= nfrag * 32) return;
    int lane = i & 31;
    int frag = i >> 5;
    int s = frag >> 2;
    int t = frag & 3;
    int hv = lane >> 4;
    int lrow = lane & 15;
    int n = t * 16 + lrow;
    _Float16* p = dst + (size_t)i * 16;
    #pragma unroll
    for (int e = 0; e < 16; ++e) {
        int k = s * 32 + hv * 8 + ((e < 8) ? e : 8 + e);
        float v = (k < half) ? srcA[k * HIDC + n] : srcB[(k - half) * HIDC + n];
        p[e] = (_Float16)v;
    }
}

// ---------------------------------------------------------------------------
// Degree: cnt[dst] += 1 per edge
// ---------------------------------------------------------------------------
__global__ void degree_kernel(const long long* __restrict__ dstI, float* __restrict__ cnt) {
    long i = (long)blockIdx.x * blockDim.x + threadIdx.x;
    long stride = (long)gridDim.x * blockDim.x;
    for (; i < NEDGES; i += stride)
        atomicAdd(&cnt[(int)dstI[i]], 1.0f);
}

// ---------------------------------------------------------------------------
// Scatter: agg[dst] += feat[src].  One thread = 4 consecutive features of one
// edge (dim is a power of two; grpShift = log2(dim/4)).
// ---------------------------------------------------------------------------
__global__ void scatter_kernel(const float* __restrict__ feat,
                               const long long* __restrict__ srcI,
                               const long long* __restrict__ dstI,
                               float* __restrict__ agg, int dim, int grpShift, long total) {
    long i = (long)blockIdx.x * blockDim.x + threadIdx.x;
    long stride = (long)gridDim.x * blockDim.x;
    int grpMask = (1 << grpShift) - 1;
    for (; i < total; i += stride) {
        int e = (int)(i >> grpShift);
        int f = ((int)i & grpMask) << 2;
        int s = (int)srcI[e];
        int d = (int)dstI[e];
        const float* fp = feat + (size_t)s * dim + f;
        float*       ap = agg  + (size_t)d * dim + f;
        atomicAdd(ap + 0, fp[0]);
        atomicAdd(ap + 1, fp[1]);
        atomicAdd(ap + 2, fp[2]);
        atomicAdd(ap + 3, fp[3]);
    }
}

// ---------------------------------------------------------------------------
// SAGE layer via WMMA:  h = relu( [agg/deg , self] @ W + bias )
// HALF = width of each source (conv1: 128, conv2: 64); K = 2*HALF.
// One wave per 16-node tile; 8 waves / 256-thread block.  B fragments come
// pre-swizzled: one aligned v16h load per (s,t) per lane.
// ---------------------------------------------------------------------------
template <int HALF>
__global__ void sage_wmma_kernel(const float* __restrict__ agg, const float* __restrict__ cnt,
                                 const float* __restrict__ self,
                                 const _Float16* __restrict__ Wsw,
                                 const float* __restrict__ bias, float* __restrict__ hout,
                                 int ntiles) {
    int wave = threadIdx.x >> 5;
    int tile = blockIdx.x * (blockDim.x >> 5) + wave;
    if (tile >= ntiles) return;                      // uniform per wave: EXEC stays all-ones
    int lane = threadIdx.x & 31;
    int half = lane >> 4;
    int lrow = lane & 15;
    int row  = tile * 16 + lrow;                     // A row owned by this lane
    float scale = 1.0f / fmaxf(cnt[row], 1.0f);
    const float* aggRow  = agg  + (size_t)row * HALF;
    const float* selfRow = self + (size_t)row * HALF;
    const v16h* Wfrag = (const v16h*)Wsw;            // [frag*32 + lane]

    v8f c[4] = {v8f{}, v8f{}, v8f{}, v8f{}};

    #pragma unroll
    for (int s = 0; s < (2 * HALF) / 32; ++s) {
        int k0 = s * 32 + half * 8;                  // e<8: k=k0+e ; e>=8: k=k0+8+e
        v16h a;
        #pragma unroll
        for (int e = 0; e < 16; ++e) {
            int k = k0 + ((e < 8) ? e : 8 + e);
            float v = (k < HALF) ? aggRow[k] * scale : selfRow[k - HALF];
            a[e] = (_Float16)v;
        }
        #pragma unroll
        for (int t = 0; t < 4; ++t) {
            v16h b = Wfrag[(s * 4 + t) * 32 + lane]; // 32B aligned, coalesced, L2-hot
            c[t] = __builtin_amdgcn_wmma_f32_16x16x32_f16(
                false, a, false, b, (short)0, c[t], false, false);
        }
    }

    #pragma unroll
    for (int t = 0; t < 4; ++t) {
        int n  = t * 16 + lrow;
        float bn = bias[n];
        #pragma unroll
        for (int r = 0; r < 8; ++r) {
            int m = tile * 16 + half * 8 + r;        // C reg r -> row r (+8 upper half-wave)
            hout[(size_t)m * HIDC + n] = fmaxf(c[t][r] + bn, 0.0f);
        }
    }
}

// ---------------------------------------------------------------------------
// Edge MLP, fused gather + WMMA GEMM1 + tiny GEMM2.
// A row e = [h2[src[e]] , h2[dst[e]]]  (edge_rep never hits HBM).
// hidden tile (16x64) staged in LDS for the 64->2 projection.
// ---------------------------------------------------------------------------
__global__ void edge_mlp_wmma_kernel(const float* __restrict__ h2,
                                     const long long* __restrict__ srcI,
                                     const long long* __restrict__ dstI,
                                     const _Float16* __restrict__ Wm1sw,
                                     const float* __restrict__ bm1,
                                     const float* __restrict__ Wm2,
                                     const float* __restrict__ bm2,
                                     float* __restrict__ out) {
    __shared__ float hidden[8][16][HIDC];            // 32 KB / block (WGP has 320 KB)
    int wave = threadIdx.x >> 5;
    int tile = blockIdx.x * (blockDim.x >> 5) + wave;   // 50000 tiles exactly
    int lane = threadIdx.x & 31;
    int half = lane >> 4;
    int lrow = lane & 15;
    int erow = tile * 16 + lrow;
    int sN = (int)srcI[erow];
    int dN = (int)dstI[erow];
    const float* rowS = h2 + (size_t)sN * HIDC;
    const float* rowD = h2 + (size_t)dN * HIDC;
    const v16h* Wfrag = (const v16h*)Wm1sw;

    v8f c[4] = {v8f{}, v8f{}, v8f{}, v8f{}};

    #pragma unroll
    for (int s = 0; s < 128 / 32; ++s) {
        int k0 = s * 32 + half * 8;
        v16h a;
        #pragma unroll
        for (int e = 0; e < 16; ++e) {
            int k = k0 + ((e < 8) ? e : 8 + e);
            float v = (k < HIDC) ? rowS[k] : rowD[k - HIDC];
            a[e] = (_Float16)v;
        }
        #pragma unroll
        for (int t = 0; t < 4; ++t) {
            v16h b = Wfrag[(s * 4 + t) * 32 + lane];
            c[t] = __builtin_amdgcn_wmma_f32_16x16x32_f16(
                false, a, false, b, (short)0, c[t], false, false);
        }
    }

    // bias + ReLU into LDS
    #pragma unroll
    for (int t = 0; t < 4; ++t) {
        int n = t * 16 + lrow;
        float bn = bm1[n];
        #pragma unroll
        for (int r = 0; r < 8; ++r)
            hidden[wave][half * 8 + r][n] = fmaxf(c[t][r] + bn, 0.0f);
    }
    __syncthreads();

    // 64 -> 2 projection: lane handles (edge m, output o)
    int m = lane >> 1;
    int o = lane & 1;
    float acc = bm2[o];
    #pragma unroll
    for (int k = 0; k < HIDC; ++k)
        acc += hidden[wave][m][k] * Wm2[k * OUTC + o];
    out[(size_t)(tile * 16 + m) * OUTC + o] = acc;
}

// ---------------------------------------------------------------------------
// Launch
// ---------------------------------------------------------------------------
extern "C" void kernel_launch(void* const* d_in, const int* in_sizes, int n_in,
                              void* d_out, int out_size, void* d_ws, size_t ws_size,
                              hipStream_t stream) {
    const float*     x    = (const float*)d_in[0];
    const long long* ei   = (const long long*)d_in[1];   // int64 [2,E]
    const float*     W1l  = (const float*)d_in[2];
    const float*     b1l  = (const float*)d_in[3];
    const float*     W1r  = (const float*)d_in[4];
    const float*     W2l  = (const float*)d_in[5];
    const float*     b2l  = (const float*)d_in[6];
    const float*     W2r  = (const float*)d_in[7];
    const float*     Wm1  = (const float*)d_in[8];
    const float*     bm1  = (const float*)d_in[9];
    const float*     Wm2  = (const float*)d_in[10];
    const float*     bm2  = (const float*)d_in[11];
    float*           out  = (float*)d_out;

    const long long* srcI = ei;
    const long long* dstI = ei + NEDGES;

    // workspace layout (agg1 | cnt | agg2 contiguous so one zero pass covers them)
    float* agg1 = (float*)d_ws;                                   // N*128
    float* cnt  = agg1 + (size_t)NNODES * IN_DIMC;                // N
    float* agg2 = cnt  + NNODES;                                  // N*64
    float* h1   = agg2 + (size_t)NNODES * HIDC;                   // N*64
    float* h2   = h1   + (size_t)NNODES * HIDC;                   // N*64
    // pre-swizzled f16 weights (fragment-lane order); h2 end is 32B aligned
    _Float16* Wc1sw = (_Float16*)(h2 + (size_t)NNODES * HIDC);    // 32 frags * 512
    _Float16* Wc2sw = Wc1sw + 32 * 32 * 16;                       // 16 frags * 512
    _Float16* Wm1sw = Wc2sw + 16 * 32 * 16;                       // 16 frags * 512

    const long zeroN = (long)NNODES * (IN_DIMC + 1 + HIDC);       // agg1+cnt+agg2

    // 1. zero accumulators (every launch: graph replays must be deterministic)
    zero_kernel<<<4096, 256, 0, stream>>>(agg1, zeroN);
    // 2. swizzle weights into WMMA fragment order (f16)
    swz_weights_kernel<<<4, 256, 0, stream>>>(W1l, W1r, Wc1sw, IN_DIMC, 8);   // 1024 threads
    swz_weights_kernel<<<2, 256, 0, stream>>>(W2l, W2r, Wc2sw, HIDC, 4);      // 512 threads
    swz_weights_kernel<<<2, 256, 0, stream>>>(Wm1, Wm1 + HIDC * HIDC, Wm1sw, HIDC, 4);
    // 3. degrees
    degree_kernel<<<2048, 256, 0, stream>>>(dstI, cnt);
    // 4. scatter x[src] -> agg1[dst]   (dim=128, shift=5)
    scatter_kernel<<<8192, 256, 0, stream>>>(x, srcI, dstI, agg1, IN_DIMC, 5,
                                             (long)NEDGES * (IN_DIMC / 4));
    // 5. conv1: [N,256] x [256,64] -> h1
    {
        int ntiles = NNODES / 16;                    // 3125
        int blocks = (ntiles + 7) / 8;
        sage_wmma_kernel<IN_DIMC><<<blocks, 256, 0, stream>>>(agg1, cnt, x, Wc1sw, b1l, h1, ntiles);
    }
    // 6. scatter h1[src] -> agg2[dst]  (dim=64, shift=4)
    scatter_kernel<<<8192, 256, 0, stream>>>(h1, srcI, dstI, agg2, HIDC, 4,
                                             (long)NEDGES * (HIDC / 4));
    // 7. conv2: [N,128] x [128,64] -> h2
    {
        int ntiles = NNODES / 16;
        int blocks = (ntiles + 7) / 8;
        sage_wmma_kernel<HIDC><<<blocks, 256, 0, stream>>>(agg2, cnt, h1, Wc2sw, b2l, h2, ntiles);
    }
    // 8. fused edge MLP -> out [E,2]
    {
        int ntiles = NEDGES / 16;                    // 50000, divisible by 8
        int blocks = ntiles / 8;                     // 6250
        edge_mlp_wmma_kernel<<<blocks, 256, 0, stream>>>(h2, srcI, dstI, Wm1sw, bm1, Wm2, bm2, out);
    }
}
